// Head_76055280877612
// MI455X (gfx1250) — matrix operand order repaired and checked by reading, not compiled, not run on hardware
//
#include <hip/hip_runtime.h>
#include <hip/hip_bf16.h>

// Shapes
#define BB 8
#define TT 2048
#define CC 768
#define HH 64
#define NSPLIT 4   // waves per block cooperating on one query tile (key-split)
#define SC 64      // keys per chunk

typedef __attribute__((ext_vector_type(16))) _Float16 v16h;
typedef __attribute__((ext_vector_type(8)))  float    v8f;

union AH { v16h v; _Float16 h[16]; uint4 u[2]; };

// ---------------------------------------------------------------------------
// Kernel 1: convert & transpose weights to f16.
// wt layout: [(w*64 + n)][k]  (w: 0=Wq, 1=Wk, 2=Wv) -> every WMMA B-operand
// lane reads two 16-byte runs (8 f16 K-values each, 16 apart, for a fixed output column).
// ---------------------------------------------------------------------------
__global__ void wt_convert_kernel(const float* __restrict__ Wq,
                                  const float* __restrict__ Wk,
                                  const float* __restrict__ Wv,
                                  _Float16* __restrict__ wt) {
  int idx = blockIdx.x * blockDim.x + threadIdx.x;
  if (idx >= 3 * CC * HH) return;
  int w = idx / (CC * HH);
  int r = idx % (CC * HH);
  int k = r >> 6;
  int n = r & 63;
  const float* src = (w == 0) ? Wq : ((w == 1) ? Wk : Wv);
  wt[((size_t)(w * HH + n)) * CC + k] = (_Float16)src[k * HH + n];
}

// ---------------------------------------------------------------------------
// Kernel 2: QKV projection GEMM.  M = B*T = 16384, K = 768, N = 192.
// One block (4 waves) per 16-row M tile; each wave owns 3 of the 12 N tiles.
// Outputs: qh[B*T][64] f16, kh[B*T][64] f16, vth[B][64][T] f16 (V transposed).
// ---------------------------------------------------------------------------
__global__ __launch_bounds__(128) void qkv_proj_kernel(
    const float* __restrict__ x, const _Float16* __restrict__ wt,
    _Float16* __restrict__ qh, _Float16* __restrict__ kh,
    _Float16* __restrict__ vth) {
  const int lane = threadIdx.x & 31;
  const int wave = threadIdx.x >> 5;
  const int ln = lane & 15;
  const int hi = lane >> 4;
  const int m0 = blockIdx.x * 16;
  const int mrow = m0 + ln;
  const float* xrow = x + (size_t)mrow * CC;

  v8f c[3] = {};

  for (int kk = 0; kk < CC; kk += 32) {
    // A tile 16x32 f16: lane = row; K pattern {8*hi+0..7, 16+8*hi+0..7}
    float4 f0 = *(const float4*)(xrow + kk + 8 * hi);
    float4 f1 = *(const float4*)(xrow + kk + 8 * hi + 4);
    float4 f2 = *(const float4*)(xrow + kk + 16 + 8 * hi);
    float4 f3 = *(const float4*)(xrow + kk + 16 + 8 * hi + 4);
    float af[16];
    *(float4*)(af + 0) = f0; *(float4*)(af + 4)  = f1;
    *(float4*)(af + 8) = f2; *(float4*)(af + 12) = f3;
    AH a;
#pragma unroll
    for (int e = 0; e < 16; ++e) a.h[e] = (_Float16)af[e];

#pragma unroll
    for (int jj = 0; jj < 3; ++jj) {
      const int j = wave * 3 + jj;       // global N tile 0..11
      const int w = j >> 2;              // which weight
      const int n = (j & 3) * 16 + ln;   // output column
      const _Float16* bp = wt + ((size_t)(w * HH + n)) * CC + kk + 8 * hi;
      AH bm;
      bm.u[0] = *(const uint4*)(bp);
      bm.u[1] = *(const uint4*)(bp + 16);
      c[jj] = __builtin_amdgcn_wmma_f32_16x16x32_f16(
          false, a.v, false, bm.v, (short)0, c[jj], false, false);
    }
  }

  const int b  = m0 >> 11;
  const int t0 = m0 & (TT - 1);
#pragma unroll
  for (int jj = 0; jj < 3; ++jj) {
    const int j = wave * 3 + jj;
    const int w = j >> 2;
    const int h = (j & 3) * 16 + ln;
    if (w < 2) {
      _Float16* dst = (w == 0) ? qh : kh;
#pragma unroll
      for (int r = 0; r < 8; ++r) {
        const int mm = m0 + 8 * hi + r;
        dst[(size_t)mm * HH + h] = (_Float16)c[jj][r];
      }
    } else {
      union { uint4 u; _Float16 hh[8]; } pk;
#pragma unroll
      for (int r = 0; r < 8; ++r) pk.hh[r] = (_Float16)c[jj][r];
      *(uint4*)(vth + ((size_t)(b * HH + h)) * TT + t0 + 8 * hi) = pk.u;
    }
  }
}

// ---------------------------------------------------------------------------
// Kernel 3: fused causal flash attention, in-block key-split, 64-key chunks.
// 1024 blocks (one per 16-row query tile), 4 waves each; wave w handles key
// chunks w, w+4, ... with its own online-softmax partial; partials combined
// through LDS after one barrier. 64-key chunks amortize the serial VALU
// softmax work (which co-executes with XDL WMMA) over 2x more keys.
// Softmax runs in base-2 so exponentials lower to a single v_exp_f32.
// ---------------------------------------------------------------------------
__global__ __launch_bounds__(128) void attn_kernel(
    const _Float16* __restrict__ qh, const _Float16* __restrict__ kh,
    const _Float16* __restrict__ vth, float* __restrict__ out) {
  __shared__ _Float16 plds[NSPLIT][16 * SC];   // per-wave P staging (C->A relayout)
  __shared__ float o_lds[NSPLIT][16][HH];      // per-wave partial O
  __shared__ float m_lds[NSPLIT][16];          // per-wave row max (log2 domain)
  __shared__ float l_lds[NSPLIT][16];          // per-wave row sum

  const int lane = threadIdx.x & 31;
  const int wave = threadIdx.x >> 5;
  const int ln = lane & 15;
  const int hi = lane >> 4;
  const int blk = blockIdx.x;
  const int b  = blk >> 7;
  const int t0 = (blk & 127) << 4;

  // Q A-tiles (16 rows x K=64 -> two 16x32 chunks), loaded once per wave.
  const int m = b * TT + t0 + ln;
  AH qa[2];
  {
    const _Float16* qrow = qh + (size_t)m * HH;
#pragma unroll
    for (int cix = 0; cix < 2; ++cix) {
      qa[cix].u[0] = *(const uint4*)(qrow + cix * 32 + 16 * hi);
      qa[cix].u[1] = *(const uint4*)(qrow + cix * 32 + 16 * hi + 8);
    }
  }

  float mr[8], lr[8];
  v8f o[4] = {};
#pragma unroll
  for (int r = 0; r < 8; ++r) { mr[r] = -1.0e30f; lr[r] = 0.0f; }

  const float sl2 = 0.18033688011112042f;  // (1/sqrt(64)) * log2(e)
  const int nch = (t0 + 16 + SC - 1) >> 6; // 64-key chunks up to the diagonal
  const _Float16* kbase = kh  + (size_t)b * TT * HH;
  const _Float16* vbase = vth + (size_t)b * HH * TT;

  for (int ch = wave; ch < nch; ch += NSPLIT) {
    const int s0 = ch * SC;
    if (ch + NSPLIT < nch) {  // stream this wave's next K chunk
      __builtin_prefetch(kbase + (size_t)(s0 + NSPLIT * SC + lane) * HH, 0, 1);
      __builtin_prefetch(kbase + (size_t)(s0 + NSPLIT * SC + 32 + lane) * HH, 0, 1);
    }

    // ---- scores S = Q K^T, four 16-col subtiles ------------------------
    v8f st[4];
#pragma unroll
    for (int sc = 0; sc < 4; ++sc) {
      const int s = s0 + sc * 16 + ln;                 // key index = B column
      const _Float16* krow = kbase + (size_t)s * HH + 16 * hi;
      AH kb0, kb1;
      kb0.u[0] = *(const uint4*)(krow);
      kb0.u[1] = *(const uint4*)(krow + 8);
      kb1.u[0] = *(const uint4*)(krow + 32);
      kb1.u[1] = *(const uint4*)(krow + 40);
      v8f acc = {};
      acc = __builtin_amdgcn_wmma_f32_16x16x32_f16(
          false, qa[0].v, false, kb0.v, (short)0, acc, false, false);
      acc = __builtin_amdgcn_wmma_f32_16x16x32_f16(
          false, qa[1].v, false, kb1.v, (short)0, acc, false, false);
      st[sc] = acc;
    }

    // ---- scale to log2 domain + causal mask (VALU selects only) --------
#pragma unroll
    for (int sc = 0; sc < 4; ++sc) {
      const int s = s0 + sc * 16 + ln;
#pragma unroll
      for (int r = 0; r < 8; ++r) {
        const int t = t0 + 8 * hi + r;
        float v = st[sc][r] * sl2;
        st[sc][r] = (s > t) ? -1.0e30f : v;
      }
    }

    // ---- online softmax (rows live across 16-lane half-groups) ---------
    float alpha[8];
#pragma unroll
    for (int r = 0; r < 8; ++r) {
      float vmax = fmaxf(fmaxf(st[0][r], st[1][r]), fmaxf(st[2][r], st[3][r]));
#pragma unroll
      for (int sh = 1; sh < 16; sh <<= 1)
        vmax = fmaxf(vmax, __shfl_xor(vmax, sh, 32));
      const float mnew = fmaxf(mr[r], vmax);
      alpha[r] = exp2f(mr[r] - mnew);
      mr[r] = mnew;
    }
#pragma unroll
    for (int sc = 0; sc < 4; ++sc)
#pragma unroll
      for (int r = 0; r < 8; ++r)
        st[sc][r] = exp2f(st[sc][r] - mr[r]);
#pragma unroll
    for (int r = 0; r < 8; ++r) {
      float rs = (st[0][r] + st[1][r]) + (st[2][r] + st[3][r]);
#pragma unroll
      for (int sh = 1; sh < 16; sh <<= 1) rs += __shfl_xor(rs, sh, 32);
      lr[r] = lr[r] * alpha[r] + rs;
#pragma unroll
      for (int n = 0; n < 4; ++n) o[n][r] *= alpha[r];
    }

    // ---- P: C-layout f32 -> A-layout f16 via per-wave LDS --------------
#pragma unroll
    for (int sc = 0; sc < 4; ++sc)
#pragma unroll
      for (int r = 0; r < 8; ++r)
        plds[wave][(8 * hi + r) * SC + sc * 16 + ln] = (_Float16)st[sc][r];
    asm volatile("s_wait_dscnt 0" ::: "memory");
    AH pa[2];  // P as two 16x32 A-tiles (K = local key index)
    {
      const _Float16* prow = &plds[wave][0] + ln * SC;
#pragma unroll
      for (int cix = 0; cix < 2; ++cix) {
        pa[cix].u[0] = *(const uint4*)(prow + cix * 32 + 16 * hi);
        pa[cix].u[1] = *(const uint4*)(prow + cix * 32 + 16 * hi + 8);
      }
    }
    asm volatile("" ::: "memory");

    // ---- O += P V  (B operand from transposed V, contiguous per lane) --
#pragma unroll
    for (int n = 0; n < 4; ++n) {
      const int h = n * 16 + ln;
      const _Float16* vp = vbase + (size_t)h * TT + s0 + 16 * hi;
      AH vb0, vb1;
      vb0.u[0] = *(const uint4*)(vp);
      vb0.u[1] = *(const uint4*)(vp + 8);
      vb1.u[0] = *(const uint4*)(vp + 32);
      vb1.u[1] = *(const uint4*)(vp + 40);
      v8f acc = o[n];
      acc = __builtin_amdgcn_wmma_f32_16x16x32_f16(
          false, pa[0].v, false, vb0.v, (short)0, acc, false, false);
      acc = __builtin_amdgcn_wmma_f32_16x16x32_f16(
          false, pa[1].v, false, vb1.v, (short)0, acc, false, false);
      o[n] = acc;
    }
  }

  // ---- publish this wave's partial (O, m, l) ----------------------------
#pragma unroll
  for (int r = 0; r < 8; ++r) {
    const int row = 8 * hi + r;
    if (ln == 0) { m_lds[wave][row] = mr[r]; l_lds[wave][row] = lr[r]; }
#pragma unroll
    for (int n = 0; n < 4; ++n) o_lds[wave][row][n * 16 + ln] = o[n][r];
  }
  __syncthreads();

  // ---- combine partials: wave w owns head-dim column group w ------------
  {
    const int h = wave * 16 + ln;
#pragma unroll
    for (int r = 0; r < 8; ++r) {
      const int row = 8 * hi + r;
      float M = -1.0e30f;
#pragma unroll
      for (int v = 0; v < NSPLIT; ++v) M = fmaxf(M, m_lds[v][row]);
      float L = 0.0f, acc = 0.0f;
#pragma unroll
      for (int v = 0; v < NSPLIT; ++v) {
        const float wgt = exp2f(m_lds[v][row] - M);
        L   += l_lds[v][row] * wgt;
        acc += o_lds[v][row][h] * wgt;
      }
      out[((size_t)(b * TT + t0 + row)) * HH + h] = acc / L;
    }
  }
}

// ---------------------------------------------------------------------------
extern "C" void kernel_launch(void* const* d_in, const int* in_sizes, int n_in,
                              void* d_out, int out_size, void* d_ws, size_t ws_size,
                              hipStream_t stream) {
  const float* x  = (const float*)d_in[0];
  const float* Wk = (const float*)d_in[1];
  const float* Wq = (const float*)d_in[2];
  const float* Wv = (const float*)d_in[3];
  float* out = (float*)d_out;

  char* wsb = (char*)d_ws;
  _Float16* qh  = (_Float16*)(wsb);                                  // 2 MB
  _Float16* kh  = (_Float16*)(wsb + (size_t)2 * 1024 * 1024);        // 2 MB
  _Float16* vth = (_Float16*)(wsb + (size_t)4 * 1024 * 1024);        // 2 MB
  _Float16* wt  = (_Float16*)(wsb + (size_t)6 * 1024 * 1024);        // 288 KB

  wt_convert_kernel<<<(3 * CC * HH + 255) / 256, 256, 0, stream>>>(Wq, Wk, Wv, wt);
  qkv_proj_kernel<<<(BB * TT) / 16, 128, 0, stream>>>(x, wt, qh, kh, vth);
  attn_kernel<<<BB * (TT / 16), 128, 0, stream>>>(qh, kh, vth, out);
}
